// Gene2SubDifferCrossAttn_42073499631974
// MI455X (gfx1250) — compile-verified, matrix operand order checked by reading, not run on hardware
//
#include <hip/hip_runtime.h>
#include <hip/hip_bf16.h>
#include <stdint.h>

// Problem constants (from reference)
#define B_   16
#define P_   32
#define G_   64
#define S_   1024
#define E_   512
#define L_   2048          // P_*G_
#define HD_  256
#define SCALING_      0.0625f   // HD_^-0.5
#define LAMBDA_INIT_  0.2f      // 0.8 - 0.6*exp(0)
#define ONE_MINUS_LI_ 0.8f
#define EPS_          1e-5f
// LDS row pitch in bf16 elems: 32 data + 8 pad. Matches TDM padding
// (pad_interval: 16 DWORDs -> code 3, pad_amount: 4 DWORDs -> code 3):
// keeps 16B alignment for ds_load_b128 and kills bank conflicts.
#define LDSP 40

typedef __attribute__((ext_vector_type(16))) __bf16       v16bf;
typedef __attribute__((ext_vector_type(8)))  float        v8f;
typedef __attribute__((ext_vector_type(4)))  unsigned int v4u;
typedef __attribute__((ext_vector_type(4)))  float        v4f;
typedef __attribute__((ext_vector_type(8)))  int          v8i;
typedef __attribute__((ext_vector_type(4)))  int          v4i;

union FragU { v16bf h; v4u u[2]; };

__device__ __forceinline__ unsigned short f32_bf16(float f) {
  unsigned int u = __float_as_uint(f);
  u += 0x7fffu + ((u >> 16) & 1u);          // round-to-nearest-even
  return (unsigned short)(u >> 16);
}
__device__ __forceinline__ unsigned int pack2(float lo, float hi) {
  return (unsigned int)f32_bf16(lo) | ((unsigned int)f32_bf16(hi) << 16);
}

// ---------------------------------------------------------------------------
// Tensor Data Mover: DMA one 2-D bf16 tile (tile_d0 elems x tile_d1 rows,
// row stride in elems) from global into LDS with 8-elem row padding (pitch
// LDSP). One instruction moves the whole tile; tracked by TENSORcnt.
// Descriptor per CDNA5 ISA (type=2, count=1,
// data_size=2B, pad 16DW->+4DW, tensor_dim==tile_dim so no OOB).
// This toolchain exposes the 6-arg builtin (g0, g1, g2, g3, g4, cpol);
// groups 2..4 are only needed for >=3-D tensors -> zero-filled.
// ---------------------------------------------------------------------------
__device__ __forceinline__ void tdm_load_2d(unsigned lds_off,
                                            const void* gptr,
                                            int tile_d0, int tile_d1,
                                            int row_stride_elems)
{
  unsigned long long ga = (unsigned long long)(uintptr_t)gptr;
  v4u g0 = { 1u,                                    // count=1
             lds_off,                               // lds_addr
             (unsigned)(ga & 0xffffffffu),          // global_addr lo
             (unsigned)((ga >> 32) & 0x1ffffffu) | 0x80000000u }; // hi | type=2
  int w0 = (1 << 16)        // data_size = 2 bytes
         | (1 << 20)        // pad_enable
         | (3 << 22)        // pad_interval: 16 DWORDs (one 32-elem bf16 row)
         | (3 << 25);       // pad_amount: 4 DWORDs (8 bf16 elems)
  int w1 = (tile_d0 & 0xffff) << 16;                       // tensor_dim0 lo16
  int w2 = ((unsigned)tile_d0 >> 16) | ((tile_d1 & 0xffff) << 16);
  int w3 = ((unsigned)tile_d1 >> 16) | ((tile_d0 & 0xffff) << 16); // tile_dim0
  int w4 = (tile_d1 & 0xffff);                             // tile_dim1 (dim2=0)
  int w5 = row_stride_elems;                               // tensor_dim0_stride
  v8i g1 = { w0, w1, w2, w3, w4, w5, 0, 0 };
  v4i z4 = { 0, 0, 0, 0 };
  v8i z8 = { 0, 0, 0, 0, 0, 0, 0, 0 };
  __builtin_amdgcn_tensor_load_to_lds(g0, g1, z4, z4, z8, 0);
}

// Load one 16x32 bf16 WMMA fragment from an LDS tile stored row-major
// (k contiguous, pitch LDSP). Per the CDNA5 16-bit A/B layout, lane group
// g=lane>>4 holds K in [8g,8g+8) and [16+8g,16+8g+8): two ds_load_b128.
__device__ __forceinline__ v16bf load_frag(const unsigned short* t, int rowBase) {
  int lane = threadIdx.x & 31;
  int r = rowBase + (lane & 15);
  int g = lane >> 4;
  FragU f;
  f.u[0] = *(const v4u*)(t + r * LDSP + 8 * g);
  f.u[1] = *(const v4u*)(t + r * LDSP + 16 + 8 * g);
  return f.h;
}

__device__ __forceinline__ v8f wmma_bf16(v16bf a, v16bf b, v8f c) {
  return __builtin_amdgcn_wmma_f32_16x16x32_bf16(false, a, false, b, (short)0, c, false, false);
}

// ---------------------------------------------------------------------------
// fp32 -> bf16 elementwise convert, 8 elems/thread (n must be multiple of 8)
// ---------------------------------------------------------------------------
__global__ __launch_bounds__(256) void cvt_k(const float* __restrict__ src,
                                             unsigned short* __restrict__ dst,
                                             int n8)
{
  int i = blockIdx.x * 256 + threadIdx.x;
  if (i >= n8) return;
  const v4f* s = (const v4f*)(src + (size_t)i * 8);
  v4f a = s[0], b = s[1];
  v4u o = { pack2(a[0], a[1]), pack2(a[2], a[3]),
            pack2(b[0], b[1]), pack2(b[2], b[3]) };
  *(v4u*)(dst + (size_t)i * 8) = o;
}

// ---------------------------------------------------------------------------
// bf16 GEMM: C[M,N] = alpha * A[M,K] * B[N,K]^T, both operands k-contiguous.
// Tile 128x64x32, 256 threads = 8 waves, each wave a 32x32 block (2x2 WMMA).
// Tiles staged by TDM, double-buffered: DMA of tile i+1 overlaps WMMA of i.
// ---------------------------------------------------------------------------
__global__ __launch_bounds__(256) void gemm_bb(
    const unsigned short* __restrict__ A, const unsigned short* __restrict__ B,
    float* __restrict__ Cf, unsigned short* __restrict__ Cb,
    int K, int lda, int ldb, int ldc,
    long long aBS, long long bBS, long long cBS, float alpha)
{
  __shared__ unsigned short At[2][128 * LDSP];   // 2 x 10 KB
  __shared__ unsigned short Bt[2][64 * LDSP];    // 2 x  5 KB

  const int tid  = threadIdx.x;
  const int lane = tid & 31;
  const int wave = tid >> 5;
  const int wm   = wave & 3;      // 0..3 -> 32-row band
  const int wn   = wave >> 2;     // 0..1 -> 32-col band
  const int z    = blockIdx.z;
  const int tm   = blockIdx.x * 128;
  const int tn   = blockIdx.y * 64;

  const unsigned short* Ab = A + (size_t)z * aBS + (size_t)tm * lda;
  const unsigned short* Bb = B + (size_t)z * bBS + (size_t)tn * ldb;

  auto stage = [&](int buf, int k0) {
    if (tid == 0) {            // wave 0 issues the DMAs once (TDM ignores EXEC)
      tdm_load_2d((unsigned)(uintptr_t)&At[buf][0], Ab + k0, 32, 128, lda);
      tdm_load_2d((unsigned)(uintptr_t)&Bt[buf][0], Bb + k0, 32, 64, ldb);
    }
  };

  v8f acc[2][2] = {};
  const int steps = K / 32;
  stage(0, 0);
  for (int i = 0; i < steps; ++i) {
    __builtin_amdgcn_s_wait_tensorcnt(0);   // wave0's DMA for buf i&1 done
    __syncthreads();                        // visible to all waves
    if (i + 1 < steps) stage((i + 1) & 1, (i + 1) * 32);

    const unsigned short* at = At[i & 1];
    const unsigned short* bt = Bt[i & 1];
    v16bf a0f = load_frag(at, wm * 32 + 0);
    v16bf a1f = load_frag(at, wm * 32 + 16);
    v16bf b0f = load_frag(bt, wn * 32 + 0);
    v16bf b1f = load_frag(bt, wn * 32 + 16);
    acc[0][0] = wmma_bf16(a0f, b0f, acc[0][0]);
    acc[0][1] = wmma_bf16(a0f, b1f, acc[0][1]);
    acc[1][0] = wmma_bf16(a1f, b0f, acc[1][0]);
    acc[1][1] = wmma_bf16(a1f, b1f, acc[1][1]);
  }

  // Epilogue. C frag layout: lane<16 -> N=lane, M=r; lane>=16 -> N=lane-16, M=r+8
  const int l16 = lane & 15, gh = lane >> 4;
  #pragma unroll
  for (int mi = 0; mi < 2; ++mi)
    #pragma unroll
    for (int ni = 0; ni < 2; ++ni)
      #pragma unroll
      for (int r = 0; r < 8; ++r) {
        int row = tm + wm * 32 + mi * 16 + gh * 8 + r;
        int col = tn + wn * 32 + ni * 16 + l16;
        float v = acc[mi][ni][r] * alpha;
        size_t idx = (size_t)z * (size_t)cBS + (size_t)row * ldc + col;
        if (Cf) Cf[idx] = v;
        if (Cb) Cb[idx] = f32_bf16(v);
      }
}

// ---------------------------------------------------------------------------
// Attention: per (b, 32-row L tile) compute both heads' scores over full
// S=1024 (TDM-staged, double-buffered across the 16 K-steps of both heads),
// softmax (mask-fill-0 semantics), write diff = a0 - lam*a1 as fp32 (output)
// and bf16 (operand of the attn@V GEMM).
// 8 waves, each owns a 128-wide S segment: 2x8 WMMA accumulators.
// ---------------------------------------------------------------------------
__global__ __launch_bounds__(256) void attn_k(
    const unsigned short* __restrict__ Qb, const unsigned short* __restrict__ Kb,
    const unsigned char* __restrict__ mask, const float* __restrict__ lamp,
    float* __restrict__ diff, unsigned short* __restrict__ diffB)
{
  __shared__ unsigned short At[2][32 * LDSP];    // 2 x 2.5 KB
  __shared__ unsigned short Bt[2][1024 * LDSP];  // 2 x 80 KB
  __shared__ float red[32 * 8];
  __shared__ float rstat[32];
  __shared__ unsigned int mrow[32];

  const int tid  = threadIdx.x;
  const int lane = tid & 31;
  const int wave = tid >> 5;
  const int l16  = lane & 15, gh = lane >> 4;
  const int sseg = wave * 128;
  const int l0   = blockIdx.x * 32;
  const int b    = blockIdx.y;
  const float lam = *lamp;

  if (tid < 32) mrow[tid] = mask[(size_t)b * L_ + l0 + tid];

  auto stage = [&](int buf, int step) {     // step = h*8 + k-chunk
    if (tid == 0) {
      int hh = step >> 3, kk = (step & 7) * 32;
      tdm_load_2d((unsigned)(uintptr_t)&At[buf][0],
                  Qb + ((size_t)(b * L_ + l0)) * E_ + hh * HD_ + kk, 32, 32, E_);
      tdm_load_2d((unsigned)(uintptr_t)&Bt[buf][0],
                  Kb + ((size_t)(b * S_)) * E_ + hh * HD_ + kk, 32, 1024, E_);
    }
  };

  stage(0, 0);
  for (int h = 0; h < 2; ++h) {
    v8f acc[2][8] = {};
    for (int kk = 0; kk < 8; ++kk) {
      const int step = h * 8 + kk;
      __builtin_amdgcn_s_wait_tensorcnt(0);
      __syncthreads();
      if (step + 1 < 16) stage((step + 1) & 1, step + 1);

      const unsigned short* at = At[step & 1];
      const unsigned short* bt = Bt[step & 1];
      v16bf af0 = load_frag(at, 0);
      v16bf af1 = load_frag(at, 16);
      #pragma unroll
      for (int ni = 0; ni < 8; ++ni) {
        v16bf bf = load_frag(bt, sseg + ni * 16);
        acc[0][ni] = wmma_bf16(af0, bf, acc[0][ni]);
        acc[1][ni] = wmma_bf16(af1, bf, acc[1][ni]);
      }
    }
    __syncthreads();

    // mask (fill 0) + per-row max over this wave's 128 cols.
    // Row m lives on one 16-lane half -> shfl_xor 1/2/4/8 reduces it.
    #pragma unroll
    for (int mi = 0; mi < 2; ++mi)
      #pragma unroll
      for (int r = 0; r < 8; ++r) {
        int m = mi * 16 + gh * 8 + r;
        float mk = mrow[m] ? 1.0f : 0.0f;
        float mx = -3.0e38f;
        #pragma unroll
        for (int ni = 0; ni < 8; ++ni) {
          float v = acc[mi][ni][r] * mk;
          acc[mi][ni][r] = v;
          mx = fmaxf(mx, v);
        }
        #pragma unroll
        for (int off = 1; off < 16; off <<= 1) mx = fmaxf(mx, __shfl_xor(mx, off, 32));
        if (l16 == 0) red[m * 8 + wave] = mx;
      }
    __syncthreads();
    if (tid < 32) {
      float mx = red[tid * 8];
      #pragma unroll
      for (int w = 1; w < 8; ++w) mx = fmaxf(mx, red[tid * 8 + w]);
      rstat[tid] = mx;
    }
    __syncthreads();

    // exp + per-row sum (exp values kept in acc)
    #pragma unroll
    for (int mi = 0; mi < 2; ++mi)
      #pragma unroll
      for (int r = 0; r < 8; ++r) {
        int m = mi * 16 + gh * 8 + r;
        float mx = rstat[m];
        float sum = 0.0f;
        #pragma unroll
        for (int ni = 0; ni < 8; ++ni) {
          float e = __expf(acc[mi][ni][r] - mx);
          acc[mi][ni][r] = e;
          sum += e;
        }
        #pragma unroll
        for (int off = 1; off < 16; off <<= 1) sum += __shfl_xor(sum, off, 32);
        if (l16 == 0) red[m * 8 + wave] = sum;
      }
    __syncthreads();
    if (tid < 32) {
      float s = 0.0f;
      #pragma unroll
      for (int w = 0; w < 8; ++w) s += red[tid * 8 + w];
      rstat[tid] = s;
    }
    __syncthreads();

    // normalize + output (head0 staged in diff fp32; head1 overwrites with
    // a0 - lam*a1 and also stores the bf16 copy for the attn@V GEMM).
    #pragma unroll
    for (int mi = 0; mi < 2; ++mi)
      #pragma unroll
      for (int r = 0; r < 8; ++r) {
        int m = mi * 16 + gh * 8 + r;
        float inv = 1.0f / rstat[m];
        #pragma unroll
        for (int ni = 0; ni < 8; ++ni) {
          float p = acc[mi][ni][r] * inv;
          size_t idx = ((size_t)(b * L_ + l0 + m)) * S_ + sseg + ni * 16 + l16;
          if (h == 0) {
            diff[idx] = p;                       // stage attn0 (same thread rereads)
          } else {
            float d = diff[idx] - lam * p;
            diff[idx]  = d;
            diffB[idx] = f32_bf16(d);
          }
        }
      }
  }
}

// RMSNorm over rows of 512, folds the (1 - lambda_init) factor; bf16 out.
__global__ __launch_bounds__(256) void rms_k(const float* __restrict__ X,
                                             unsigned short* __restrict__ Y,
                                             const float* __restrict__ w)
{
  __shared__ float s8[8];
  const int tid = threadIdx.x;
  const float* x = X + (size_t)blockIdx.x * E_;
  unsigned short* y = Y + (size_t)blockIdx.x * E_;
  float a = x[tid], c = x[tid + 256];
  float ss = a * a + c * c;
  #pragma unroll
  for (int off = 1; off < 32; off <<= 1) ss += __shfl_xor(ss, off, 32);
  if ((tid & 31) == 0) s8[tid >> 5] = ss;
  __syncthreads();
  float tot = 0.0f;
  #pragma unroll
  for (int i = 0; i < 8; ++i) tot += s8[i];
  float r = rsqrtf(tot * (1.0f / E_) + EPS_) * ONE_MINUS_LI_;
  y[tid]       = f32_bf16(a * r * w[tid]);
  y[tid + 256] = f32_bf16(c * r * w[tid + 256]);
}

// lam = exp(sum lq1*lk1) - exp(sum lq2*lk2) + lambda_init  (HD_=256 elems)
__global__ __launch_bounds__(256) void lam_k(
    const float* __restrict__ lq1, const float* __restrict__ lk1,
    const float* __restrict__ lq2, const float* __restrict__ lk2,
    float* __restrict__ out)
{
  __shared__ float s1[8], s2[8];
  const int tid = threadIdx.x;
  float p1 = lq1[tid] * lk1[tid];
  float p2 = lq2[tid] * lk2[tid];
  #pragma unroll
  for (int off = 1; off < 32; off <<= 1) {
    p1 += __shfl_xor(p1, off, 32);
    p2 += __shfl_xor(p2, off, 32);
  }
  if ((tid & 31) == 0) { s1[tid >> 5] = p1; s2[tid >> 5] = p2; }
  __syncthreads();
  if (tid == 0) {
    float a = 0.0f, c = 0.0f;
    #pragma unroll
    for (int i = 0; i < 8; ++i) { a += s1[i]; c += s2[i]; }
    out[0] = __expf(a) - __expf(c) + LAMBDA_INIT_;
  }
}

extern "C" void kernel_launch(void* const* d_in, const int* in_sizes, int n_in,
                              void* d_out, int out_size, void* d_ws, size_t ws_size,
                              hipStream_t stream)
{
  const float* query = (const float*)d_in[0];
  const float* key   = (const float*)d_in[1];
  const unsigned char* mask = (const unsigned char*)d_in[2];
  const float* Wq  = (const float*)d_in[3];
  const float* Wk  = (const float*)d_in[4];
  const float* Wv  = (const float*)d_in[5];
  const float* Wo  = (const float*)d_in[6];
  const float* lq1 = (const float*)d_in[7];
  const float* lk1 = (const float*)d_in[8];
  const float* lq2 = (const float*)d_in[9];
  const float* lk2 = (const float*)d_in[10];
  const float* rms_w = (const float*)d_in[11];

  float* out  = (float*)d_out;                        // [B,L,E]
  float* diff = out + (size_t)B_ * L_ * E_;           // [B,L,S]

  // Workspace layout. Region A (qxb,kxb,Qb,Kb: 100.66 MB) is dead after
  // attn_k and is re-used for AO (fp32) + AOb (bf16): exactly 100.66 MB.
  unsigned short* qxb = (unsigned short*)d_ws;             // query bf16  [B*L,E]
  unsigned short* kxb = qxb + (size_t)B_ * L_ * E_;        // key   bf16  [B*S,E]
  unsigned short* Qb  = kxb + (size_t)B_ * S_ * E_;        // Q bf16 (pre-scaled)
  unsigned short* Kb  = Qb  + (size_t)B_ * L_ * E_;        // K bf16
  unsigned short* diffB = Kb + (size_t)B_ * S_ * E_;       // diff bf16 [B*L,S]
  unsigned short* Vt  = diffB + (size_t)B_ * L_ * S_;      // V^T bf16 [B,E,S]
  unsigned short* wqb = Vt + (size_t)B_ * E_ * S_;         // weights bf16
  unsigned short* wkb = wqb + E_ * E_;
  unsigned short* wvb = wkb + E_ * E_;
  unsigned short* wob = wvb + E_ * E_;
  float* lam = (float*)(wob + E_ * E_);
  float*          AO  = (float*)qxb;                       // attn_out fp32 [B*L,E]
  unsigned short* AOb = (unsigned short*)((char*)qxb + (size_t)B_ * L_ * E_ * 4);

  dim3 blk(256);

  lam_k<<<1, blk, 0, stream>>>(lq1, lk1, lq2, lk2, lam);

  // one-time fp32 -> bf16 conversions (bandwidth-trivial vs. the GEMMs)
  cvt_k<<<dim3((B_ * L_ * E_) / 8 / 256), blk, 0, stream>>>(query, qxb, B_ * L_ * E_ / 8);
  cvt_k<<<dim3((B_ * S_ * E_) / 8 / 256), blk, 0, stream>>>(key, kxb, B_ * S_ * E_ / 8);
  cvt_k<<<dim3((E_ * E_) / 8 / 256), blk, 0, stream>>>(Wq, wqb, E_ * E_ / 8);
  cvt_k<<<dim3((E_ * E_) / 8 / 256), blk, 0, stream>>>(Wk, wkb, E_ * E_ / 8);
  cvt_k<<<dim3((E_ * E_) / 8 / 256), blk, 0, stream>>>(Wv, wvb, E_ * E_ / 8);
  cvt_k<<<dim3((E_ * E_) / 8 / 256), blk, 0, stream>>>(Wo, wob, E_ * E_ / 8);

  // Q = scale * query @ Wq^T
  gemm_bb<<<dim3((B_ * L_) / 128, E_ / 64, 1), blk, 0, stream>>>(
      qxb, wqb, nullptr, Qb, E_, E_, E_, E_, 0, 0, 0, SCALING_);

  // K = key @ Wk^T
  gemm_bb<<<dim3((B_ * S_) / 128, E_ / 64, 1), blk, 0, stream>>>(
      kxb, wkb, nullptr, Kb, E_, E_, E_, E_, 0, 0, 0, 1.0f);

  // V^T = Wv @ key^T -> [B,E,S] (so the attn@V GEMM is k-contiguous)
  gemm_bb<<<dim3(E_ / 128, S_ / 64, B_), blk, 0, stream>>>(
      wvb, kxb, nullptr, Vt, E_, E_, E_, S_,
      0, (long long)S_ * E_, (long long)E_ * S_, 1.0f);

  // diff attention weights -> d_out (fp32) + diffB (bf16)
  attn_k<<<dim3(L_ / 32, B_), blk, 0, stream>>>(Qb, Kb, mask, lam, diff, diffB);

  // attn_out = diff @ V
  gemm_bb<<<dim3(L_ / 128, E_ / 64, B_), blk, 0, stream>>>(
      diffB, Vt, AO, nullptr, S_, S_, S_, E_,
      (long long)L_ * S_, (long long)E_ * S_, (long long)L_ * E_, 1.0f);

  // RMSNorm (affine) * (1 - lambda_init) -> bf16
  rms_k<<<dim3(B_ * L_), blk, 0, stream>>>(AO, AOb, rms_w);

  // out = attn_norm @ Wo^T -> fp32 d_out
  gemm_bb<<<dim3((B_ * L_) / 128, E_ / 64, 1), blk, 0, stream>>>(
      AOb, wob, out, nullptr, E_, E_, E_, E_, 0, 0, 0, 1.0f);
}
